// RetinaNet_10282151707192
// MI455X (gfx1250) — matrix operand order repaired and checked by reading, not compiled, not run on hardware
//
#include <hip/hip_runtime.h>
#include <hip/hip_bf16.h>
#include <math.h>

typedef __bf16 bf16_t;
typedef __attribute__((ext_vector_type(16))) __bf16 v16bf;
typedef __attribute__((ext_vector_type(8)))  __bf16 v8bf;
typedef __attribute__((ext_vector_type(8)))  float  v8f;

#define IN_C   256
#define KTOT   2304          // 9 * 256
#define NCLS   91
#define NA     9
#define OPAD_CLS 832         // 819 padded to /64
#define OPAD_REG 64          // 36 padded to /64
#define BBOX_CLAMP 4.135166556742356f   // ln(1000/16)

// flat shared pointer -> LDS byte offset (aperture maps addr[31:0] to LDS)
__device__ __forceinline__ uint32_t lds_off(const void* p) {
    return (uint32_t)(uintptr_t)p;
}

// ---------------------------------------------------------------------------
// Weight transform: [L][O][I][3][3] f32  ->  [L][O][2304] bf16, k = tap*256+ic
// ---------------------------------------------------------------------------
__global__ void xform_tower_w(const float* __restrict__ src, bf16_t* __restrict__ dst)
{
    size_t idx = (size_t)blockIdx.x * 256 + threadIdx.x;
    const size_t total = (size_t)4 * 256 * KTOT;
    if (idx >= total) return;
    int k = (int)(idx % KTOT);
    size_t lo = idx / KTOT;
    int o = (int)(lo % 256);
    int l = (int)(lo / 256);
    int p = k >> 8;
    int i = k & 255;
    dst[idx] = (bf16_t)src[(((size_t)(l * 256 + o) * 256 + i) * 9) + p];
}

__global__ void xform_out_w(const float* __restrict__ src, bf16_t* __restrict__ dst,
                            int O_real, int Opad)
{
    size_t idx = (size_t)blockIdx.x * 256 + threadIdx.x;
    const size_t total = (size_t)Opad * KTOT;
    if (idx >= total) return;
    int k = (int)(idx % KTOT);
    int o = (int)(idx / KTOT);
    int p = k >> 8;
    int i = k & 255;
    float v = 0.f;
    if (o < O_real) v = src[((size_t)o * 256 + i) * 9 + p];
    dst[idx] = (bf16_t)v;
}

// ---------------------------------------------------------------------------
// NCHW f32 feature -> padded HWC bf16  [(H+2)][(W+2)][256], zero border
// ---------------------------------------------------------------------------
__global__ void pad_hwc(const float* __restrict__ feat, bf16_t* __restrict__ dst,
                        int H, int W)
{
    int Wp = W + 2, Hp = H + 2;
    size_t total = (size_t)Hp * Wp * IN_C;
    size_t idx = (size_t)blockIdx.x * 256 + threadIdx.x;
    if (idx >= total) return;
    int c = (int)(idx % IN_C);
    size_t q = idx / IN_C;
    int px = (int)(q % Wp);
    int py = (int)(q / Wp);
    float v = 0.f;
    if (py >= 1 && py <= H && px >= 1 && px <= W)
        v = feat[(size_t)c * H * W + (size_t)(py - 1) * W + (px - 1)];
    dst[idx] = (bf16_t)v;
}

__global__ void zero_bf16(bf16_t* __restrict__ p, size_t n)
{
    size_t idx = (size_t)blockIdx.x * 256 + threadIdx.x;
    if (idx < n) p[idx] = (bf16_t)0.f;
}

// ---------------------------------------------------------------------------
// 3x3 conv, Cin=256, implicit GEMM on v_wmma_f32_16x16x32_bf16 with
// double-buffered LDS tiles filled by GLOBAL_LOAD_ASYNC_TO_LDS_B128.
//
// Block = 256 thr = 8 waves (4 M x 2 N) -> tile 128 px x 64 oc.
// Each wave: 2x2 fragments of 16x16 -> 8 WMMA per K-chunk of 64.
// K loop: 36 chunks of 64 (9 taps x 4 ic-chunks).
// LDS: A[2][128][64] bf16 (16KB x2) + B[2][64][64] bf16 (8KB x2) = 48KB.
// ---------------------------------------------------------------------------
template<bool RELU, bool TO_BF16>
__global__ __launch_bounds__(256)
void conv3x3_wmma(const bf16_t* __restrict__ X,
                  const bf16_t* __restrict__ Wt,
                  const float*  __restrict__ bias, int O_real,
                  void* __restrict__ out,
                  int H, int W, int Opad)
{
    __shared__ bf16_t shA[2][128 * 64];
    __shared__ bf16_t shB[2][64 * 64];

    const int tid  = threadIdx.x;
    const int lane = tid & 31;
    const int wid  = tid >> 5;
    const int mw   = wid & 3;          // M-wave: 0..3  (32 px each)
    const int nw   = wid >> 2;         // N-wave: 0..1  (32 oc each)
    const int half = lane >> 4;
    const int l16  = lane & 15;
    const int HW   = H * W;
    const int Wp   = W + 2;

    // ---- staging assignment: 1024 A-segs (4/thread) + 512 B-segs (2/thread)
    const int segp = tid >> 3;              // 0..31 : base pixel / weight row
    const int sege = (tid & 7) * 8;         // element offset inside 64-elem row
    int py[4], px[4];
#pragma unroll
    for (int j = 0; j < 4; ++j) {
        int p = blockIdx.x * 128 + j * 32 + segp;
        if (p >= HW) p = HW - 1;
        py[j] = p / W;
        px[j] = p - py[j] * W;
    }
    const bf16_t* wB0 = Wt + (size_t)(blockIdx.y * 64 + segp) * KTOT + sege;
    const bf16_t* wB1 = Wt + (size_t)(blockIdx.y * 64 + 32 + segp) * KTOT + sege;

    uint32_t dA[4];
#pragma unroll
    for (int j = 0; j < 4; ++j)
        dA[j] = lds_off(&shA[0][(j * 32 + segp) * 64 + sege]);
    const uint32_t dB0 = lds_off(&shB[0][segp * 64 + sege]);
    const uint32_t dB1 = lds_off(&shB[0][(32 + segp) * 64 + sege]);
    const uint32_t strA = 128 * 64 * (uint32_t)sizeof(bf16_t);
    const uint32_t strB = 64 * 64 * (uint32_t)sizeof(bf16_t);

    auto stage = [&](int c, int b) {
        const int pos = c >> 2;
        const int ic0 = (c & 3) << 6;
        const int dy = pos / 3;
        const int dx = pos - dy * 3;
#pragma unroll
        for (int j = 0; j < 4; ++j) {
            const uint64_t g = (uint64_t)(X + ((size_t)(py[j] + dy) * Wp + (px[j] + dx)) * IN_C + ic0 + sege);
            const uint32_t d = dA[j] + (uint32_t)b * strA;
            asm volatile("global_load_async_to_lds_b128 %0, %1, off" :: "v"(d), "v"(g) : "memory");
        }
        {
            const uint64_t g = (uint64_t)(wB0 + (size_t)c * 64);
            const uint32_t d = dB0 + (uint32_t)b * strB;
            asm volatile("global_load_async_to_lds_b128 %0, %1, off" :: "v"(d), "v"(g) : "memory");
        }
        {
            const uint64_t g = (uint64_t)(wB1 + (size_t)c * 64);
            const uint32_t d = dB1 + (uint32_t)b * strB;
            asm volatile("global_load_async_to_lds_b128 %0, %1, off" :: "v"(d), "v"(g) : "memory");
        }
    };

    // ---- fragment LDS row bases (elements, rows are 64 wide) ----
    const int rowA0 = (mw * 32 + l16) * 64;
    const int rowA1 = rowA0 + 16 * 64;
    const int rowB0 = (nw * 32 + l16) * 64;
    const int rowB1 = rowB0 + 16 * 64;
    const int ha = 8 * half;     // A lane-half k offset
    const int hb = 8 * half;     // B lane-half k offset

    v8f acc00 = {}, acc01 = {}, acc10 = {}, acc11 = {};

    stage(0, 0);
    for (int c = 0; c < 36; ++c) {
        const int b = c & 1;
        if (c < 35) {
            stage(c + 1, b ^ 1);
            asm volatile("s_wait_asynccnt 0x6" ::: "memory");  // chunk c resident
        } else {
            asm volatile("s_wait_asynccnt 0x0" ::: "memory");
        }
        __syncthreads();

        const bf16_t* As = shA[b];
        const bf16_t* Bs = shB[b];
#pragma unroll
        for (int s = 0; s < 2; ++s) {
            const int ka = s * 32 + ha;
            const int kb = s * 32 + hb;
            v8bf a00 = *(const v8bf*)(As + rowA0 + ka);
            v8bf a01 = *(const v8bf*)(As + rowA0 + ka + 16);
            v8bf a10 = *(const v8bf*)(As + rowA1 + ka);
            v8bf a11 = *(const v8bf*)(As + rowA1 + ka + 16);
            v8bf b00 = *(const v8bf*)(Bs + rowB0 + kb);
            v8bf b01 = *(const v8bf*)(Bs + rowB0 + kb + 16);
            v8bf b10 = *(const v8bf*)(Bs + rowB1 + kb);
            v8bf b11 = *(const v8bf*)(Bs + rowB1 + kb + 16);
            v16bf A0 = __builtin_shufflevector(a00, a01, 0,1,2,3,4,5,6,7,8,9,10,11,12,13,14,15);
            v16bf A1 = __builtin_shufflevector(a10, a11, 0,1,2,3,4,5,6,7,8,9,10,11,12,13,14,15);
            v16bf B0 = __builtin_shufflevector(b00, b01, 0,1,2,3,4,5,6,7,8,9,10,11,12,13,14,15);
            v16bf B1 = __builtin_shufflevector(b10, b11, 0,1,2,3,4,5,6,7,8,9,10,11,12,13,14,15);

            acc00 = __builtin_amdgcn_wmma_f32_16x16x32_bf16(false, A0, false, B0, (short)0, acc00, false, false);
            acc01 = __builtin_amdgcn_wmma_f32_16x16x32_bf16(false, A0, false, B1, (short)0, acc01, false, false);
            acc10 = __builtin_amdgcn_wmma_f32_16x16x32_bf16(false, A1, false, B0, (short)0, acc10, false, false);
            acc11 = __builtin_amdgcn_wmma_f32_16x16x32_bf16(false, A1, false, B1, (short)0, acc11, false, false);
        }
        __syncthreads();
    }

    // ---- epilogue ----
    const int n0 = blockIdx.y * 64 + nw * 32 + l16;        // frag col 0
    const int n1 = n0 + 16;                                // frag col 1
    float bv0 = 0.f, bv1 = 0.f;
    if (bias != nullptr) {
        if (n0 < O_real) bv0 = bias[n0];
        if (n1 < O_real) bv1 = bias[n1];
    }
    const int m_base = blockIdx.x * 128 + mw * 32;

#pragma unroll
    for (int r = 0; r < 8; ++r) {
        int rr = r + 8 * half;
        int p0 = m_base + rr;          // frag row 0
        int p1 = m_base + 16 + rr;     // frag row 1
        float v00 = acc00[r] + bv0, v01 = acc01[r] + bv1;
        float v10 = acc10[r] + bv0, v11 = acc11[r] + bv1;
        if (RELU) {
            v00 = fmaxf(v00, 0.f); v01 = fmaxf(v01, 0.f);
            v10 = fmaxf(v10, 0.f); v11 = fmaxf(v11, 0.f);
        }
        if (TO_BF16) {
            bf16_t* o = (bf16_t*)out;
            if (p0 < HW) {
                int yy = p0 / W, xx = p0 - (p0 / W) * W;
                size_t base = ((size_t)(yy + 1) * Wp + (xx + 1)) * IN_C;
                o[base + n0] = (bf16_t)v00;
                o[base + n1] = (bf16_t)v01;
            }
            if (p1 < HW) {
                int yy = p1 / W, xx = p1 - (p1 / W) * W;
                size_t base = ((size_t)(yy + 1) * Wp + (xx + 1)) * IN_C;
                o[base + n0] = (bf16_t)v10;
                o[base + n1] = (bf16_t)v11;
            }
        } else {
            float* o = (float*)out;
            if (p0 < HW) { o[(size_t)p0 * Opad + n0] = v00; o[(size_t)p0 * Opad + n1] = v01; }
            if (p1 < HW) { o[(size_t)p1 * Opad + n0] = v10; o[(size_t)p1 * Opad + n1] = v11; }
        }
    }
}

// ---------------------------------------------------------------------------
// Sigmoid + torchvision box decode -> out[row][95]
// ---------------------------------------------------------------------------
__global__ void postprocess(const float* __restrict__ zc,
                            const float* __restrict__ zr,
                            float* __restrict__ out,
                            int H, int W, int stride, float base_size, int row_off)
{
    int idx = blockIdx.x * 256 + threadIdx.x;      // = p*9 + a
    int HW = H * W;
    if (idx >= HW * NA) return;
    int a = idx % NA;
    int p = idx / NA;
    int y = p / W;
    int x = p - y * W;

    int ri = a / 3, si = a - ri * 3;
    float size = base_size * exp2f((float)si * (1.f / 3.f));
    float ratio = (ri == 0) ? 0.5f : (ri == 1 ? 1.0f : 2.0f);
    float hr = sqrtf(ratio);
    float aw = (1.f / hr) * size;
    float ah = hr * size;
    float acx = (float)x * (float)stride;
    float acy = (float)y * (float)stride;

    const float* r = zr + (size_t)p * OPAD_REG + a * 4;
    float dx = r[0], dy = r[1];
    float dw = fminf(r[2], BBOX_CLAMP);
    float dh = fminf(r[3], BBOX_CLAMP);
    float pcx = dx * aw + acx;
    float pcy = dy * ah + acy;
    float pw  = expf(dw) * aw;
    float ph  = expf(dh) * ah;

    float* o = out + (size_t)(row_off + idx) * 95;
    const float* c = zc + (size_t)p * OPAD_CLS + a * NCLS;
#pragma unroll 7
    for (int j = 0; j < NCLS; ++j)
        o[j] = 1.f / (1.f + expf(-c[j]));
    o[91] = pcx - 0.5f * pw;
    o[92] = pcy - 0.5f * ph;
    o[93] = pcx + 0.5f * pw;
    o[94] = pcy + 0.5f * ph;
}

// ---------------------------------------------------------------------------
// Orchestration
// ---------------------------------------------------------------------------
static const int   LVL_H[5] = {100, 50, 25, 13, 7};
static const int   LVL_W[5] = {100, 50, 25, 13, 7};
static const int   LVL_S[5] = {8, 16, 32, 64, 128};
static const float LVL_B[5] = {32.f, 64.f, 128.f, 256.f, 512.f};

extern "C" void kernel_launch(void* const* d_in, const int* in_sizes, int n_in,
                              void* d_out, int out_size, void* d_ws, size_t ws_size,
                              hipStream_t stream)
{
    (void)in_sizes; (void)n_in; (void)out_size; (void)ws_size;
    const float* feats[5] = {(const float*)d_in[0], (const float*)d_in[1],
                             (const float*)d_in[2], (const float*)d_in[3],
                             (const float*)d_in[4]};
    const float* cls_tw    = (const float*)d_in[5];
    const float* cls_tb    = (const float*)d_in[6];
    const float* cls_out_w = (const float*)d_in[7];
    const float* cls_out_b = (const float*)d_in[8];
    const float* reg_tw    = (const float*)d_in[9];
    const float* reg_tb    = (const float*)d_in[10];
    const float* reg_out_w = (const float*)d_in[11];
    const float* reg_out_b = (const float*)d_in[12];
    float* out = (float*)d_out;

    char*  ws  = (char*)d_ws;
    size_t off = 0;
    auto carve = [&](size_t bytes) -> void* {
        void* p = ws + off;
        off += (bytes + 255) & ~(size_t)255;
        return p;
    };
    bf16_t* wtc   = (bf16_t*)carve((size_t)4 * 256 * KTOT * sizeof(bf16_t));
    bf16_t* wtr   = (bf16_t*)carve((size_t)4 * 256 * KTOT * sizeof(bf16_t));
    bf16_t* wco   = (bf16_t*)carve((size_t)OPAD_CLS * KTOT * sizeof(bf16_t));
    bf16_t* wro   = (bf16_t*)carve((size_t)OPAD_REG * KTOT * sizeof(bf16_t));
    bf16_t* bufA  = (bf16_t*)carve((size_t)102 * 102 * IN_C * sizeof(bf16_t));
    bf16_t* bufB  = (bf16_t*)carve((size_t)102 * 102 * IN_C * sizeof(bf16_t));
    float*  z_cls = (float*)carve((size_t)10000 * OPAD_CLS * sizeof(float));
    float*  z_reg = (float*)carve((size_t)10000 * OPAD_REG * sizeof(float));

    {
        size_t n = (size_t)4 * 256 * KTOT;
        xform_tower_w<<<dim3((unsigned)((n + 255) / 256)), dim3(256), 0, stream>>>(cls_tw, wtc);
        xform_tower_w<<<dim3((unsigned)((n + 255) / 256)), dim3(256), 0, stream>>>(reg_tw, wtr);
        size_t nc = (size_t)OPAD_CLS * KTOT;
        xform_out_w<<<dim3((unsigned)((nc + 255) / 256)), dim3(256), 0, stream>>>(
            cls_out_w, wco, NA * NCLS, OPAD_CLS);
        size_t nr = (size_t)OPAD_REG * KTOT;
        xform_out_w<<<dim3((unsigned)((nr + 255) / 256)), dim3(256), 0, stream>>>(
            reg_out_w, wro, NA * 4, OPAD_REG);
    }

    int row_off = 0;
    for (int l = 0; l < 5; ++l) {
        const int H = LVL_H[l], W = LVL_W[l];
        const int HW = H * W;
        const int Hp = H + 2, Wp = W + 2;
        const size_t padN = (size_t)Hp * Wp * IN_C;
        const unsigned padBlocks = (unsigned)((padN + 255) / 256);
        const unsigned mTiles    = (unsigned)((HW + 127) / 128);

        zero_bf16<<<dim3(padBlocks), dim3(256), 0, stream>>>(bufB, padN);

        // ---- classification head ----
        pad_hwc<<<dim3(padBlocks), dim3(256), 0, stream>>>(feats[l], bufA, H, W);
        conv3x3_wmma<true, true><<<dim3(mTiles, IN_C / 64), dim3(256), 0, stream>>>(
            bufA, wtc + (size_t)0 * 256 * KTOT, cls_tb + 0 * 256, 256, bufB, H, W, IN_C);
        conv3x3_wmma<true, true><<<dim3(mTiles, IN_C / 64), dim3(256), 0, stream>>>(
            bufB, wtc + (size_t)1 * 256 * KTOT, cls_tb + 1 * 256, 256, bufA, H, W, IN_C);
        conv3x3_wmma<true, true><<<dim3(mTiles, IN_C / 64), dim3(256), 0, stream>>>(
            bufA, wtc + (size_t)2 * 256 * KTOT, cls_tb + 2 * 256, 256, bufB, H, W, IN_C);
        conv3x3_wmma<true, true><<<dim3(mTiles, IN_C / 64), dim3(256), 0, stream>>>(
            bufB, wtc + (size_t)3 * 256 * KTOT, cls_tb + 3 * 256, 256, bufA, H, W, IN_C);
        conv3x3_wmma<false, false><<<dim3(mTiles, OPAD_CLS / 64), dim3(256), 0, stream>>>(
            bufA, wco, cls_out_b, NA * NCLS, z_cls, H, W, OPAD_CLS);

        // ---- regression head ----
        pad_hwc<<<dim3(padBlocks), dim3(256), 0, stream>>>(feats[l], bufA, H, W);
        conv3x3_wmma<true, true><<<dim3(mTiles, IN_C / 64), dim3(256), 0, stream>>>(
            bufA, wtr + (size_t)0 * 256 * KTOT, reg_tb + 0 * 256, 256, bufB, H, W, IN_C);
        conv3x3_wmma<true, true><<<dim3(mTiles, IN_C / 64), dim3(256), 0, stream>>>(
            bufB, wtr + (size_t)1 * 256 * KTOT, reg_tb + 1 * 256, 256, bufA, H, W, IN_C);
        conv3x3_wmma<true, true><<<dim3(mTiles, IN_C / 64), dim3(256), 0, stream>>>(
            bufA, wtr + (size_t)2 * 256 * KTOT, reg_tb + 2 * 256, 256, bufB, H, W, IN_C);
        conv3x3_wmma<true, true><<<dim3(mTiles, IN_C / 64), dim3(256), 0, stream>>>(
            bufB, wtr + (size_t)3 * 256 * KTOT, reg_tb + 3 * 256, 256, bufA, H, W, IN_C);
        conv3x3_wmma<false, false><<<dim3(mTiles, OPAD_REG / 64), dim3(256), 0, stream>>>(
            bufA, wro, reg_out_b, NA * 4, z_reg, H, W, OPAD_REG);

        unsigned ppBlocks = (unsigned)(((size_t)HW * NA + 255) / 256);
        postprocess<<<dim3(ppBlocks), dim3(256), 0, stream>>>(
            z_cls, z_reg, out, H, W, LVL_S[l], LVL_B[l], row_off);
        row_off += HW * NA;
    }
}
